// SparseRegisterStep_71347996721408
// MI455X (gfx1250) — matrix operand order repaired and checked by reading, not compiled, not run on hardware
//
#include <hip/hip_runtime.h>

#define B_     2
#define T_     2048
#define V_     32000
#define K_     256
#define INNER_ 512
#define BT_    (B_*T_)
#define EPS_   1.1920929e-07f

typedef _Float16 h16;
typedef __attribute__((ext_vector_type(16))) _Float16 v16h;
typedef __attribute__((ext_vector_type(8)))  _Float16 v8h;
typedef __attribute__((ext_vector_type(8)))  float    v8f;
typedef __attribute__((ext_vector_type(4)))  float    f32x4;
typedef __attribute__((ext_vector_type(4)))  unsigned int u32x4;
typedef __attribute__((ext_vector_type(8)))  int      i32x8;
typedef __attribute__((ext_vector_type(4)))  int      i32x4;

#define WMMA(a,b,c) __builtin_amdgcn_wmma_f32_16x16x32_f16(false,(a),false,(b),(short)0,(c),false,false)

// A-matrix 16x32 f16 tile load in contiguous k order (k = 16*hs + s), matching loadB:
// lane m (=lane&15), hs (=lane>>4): halfs [c0+hs*16 .. +15] -> a[0..15]
__device__ __forceinline__ v16h loadA(const h16* tile_base, int ld, int m, int hs, int c0) {
  const h16* p = tile_base + (size_t)m * ld + c0 + hs * 16;
  v8h lo = *(const v8h*)p;
  v8h hi = *(const v8h*)(p + 8);
  return __builtin_shufflevector(lo, hi, 0,1,2,3,4,5,6,7,8,9,10,11,12,13,14,15);
}
// B-matrix 32x16 f16 tile: lane holds column n, rows k0..k0+15 contiguous in memory
// (works for global *and* LDS pointers; LDS case lowers to two ds_load_b128)
__device__ __forceinline__ v16h loadB(const h16* p) { return *(const v16h*)p; }

// ---- Tensor Data Mover: stage one contiguous 32x256 f16 k-chunk into LDS ----
// D# per CDNA5 ISA ch.8: group0 = {count|flags, lds_addr, global_addr_lo,
// global_addr_hi|type=2}; group1 packs data_size=2B, tensor/tile dims, strides.
// clang-23 toolchain declares the 6-arg builtin: (v4u g0, v8i g1, v4i, v4i, v8i, i32 cpol)
__device__ __forceinline__ void tdm_load_k(const h16* gptr, unsigned lds_off) {
  unsigned long long ga = (unsigned long long)(uintptr_t)gptr;
  u32x4 g0;
  g0.x = 1u;                                            // count=1, user descriptor
  g0.y = lds_off;                                       // LDS byte address
  g0.z = (unsigned)(ga & 0xffffffffu);                  // global_addr[31:0]
  g0.w = (unsigned)((ga >> 32) & 0x1ffffffu) | (2u << 30); // [56:32] | type=2
  // data_size=1(2B); tensor_dim0=256; tensor_dim1=1<<20 (no clip);
  // tile_dim0=256; tile_dim1=32; dim0_stride=256; dim1_stride=256
  i32x8 g1 = { 0x00010000, 0x01000000, 0, 0x01000010, 32, 256, 0x01000000, 0 };
  i32x4 gz4 = { 0, 0, 0, 0 };                           // groups 2/3: 2D tile -> unused
  i32x8 gz8 = { 0, 0, 0, 0, 0, 0, 0, 0 };
  __builtin_amdgcn_tensor_load_to_lds(g0, g1, gz4, gz4, gz8, 0);
}

// ---------------- zero output (bandwidth-critical: ~524 MB) ----------------
// Non-temporal: stream past WGP$/L2 so the 192MB L2 keeps the working set.
__global__ void k_zero(f32x4* __restrict__ out, size_t n4) {
  size_t i = (size_t)blockIdx.x * blockDim.x + threadIdx.x;
  size_t stride = (size_t)gridDim.x * blockDim.x;
  f32x4 z = { 0.f, 0.f, 0.f, 0.f };
  for (; i < n4; i += stride) __builtin_nontemporal_store(z, &out[i]);
}

// ---------------- f32 -> f16 convert (weights) ----------------
__global__ void k_f2h(const float* __restrict__ s, h16* __restrict__ d, int n) {
  int i = blockIdx.x * blockDim.x + threadIdx.x;
  int stride = gridDim.x * blockDim.x;
  for (; i < n; i += stride) d[i] = (h16)s[i];
}

// ---------------- gather read_indices + RMS norm ----------------
__global__ void k_gather_rms(const float* __restrict__ x, const int* __restrict__ ridx,
                             float* __restrict__ g, h16* __restrict__ gn) {
  int row = blockIdx.x;          // b*T + t
  int c = threadIdx.x;           // 0..255
  float val = x[(size_t)row * V_ + ridx[c]];
  g[(size_t)row * K_ + c] = val;
  __shared__ float red[256];
  red[c] = val * val;
  __syncthreads();
  for (int o = 128; o > 0; o >>= 1) {
    if (c < o) red[c] += red[c + o];
    __syncthreads();
  }
  float r = rsqrtf(red[0] * (1.f / K_) + EPS_);
  gn[(size_t)row * K_ + c] = (h16)(val * r);
}

// ---------------- RMS norm (f32 in, f16 out) ----------------
__global__ void k_rms(const float* __restrict__ src, h16* __restrict__ dst) {
  int row = blockIdx.x;
  int c = threadIdx.x;
  float val = src[(size_t)row * K_ + c];
  __shared__ float red[256];
  red[c] = val * val;
  __syncthreads();
  for (int o = 128; o > 0; o >>= 1) {
    if (c < o) red[c] += red[c + o];
    __syncthreads();
  }
  float r = rsqrtf(red[0] * (1.f / K_) + EPS_);
  dst[(size_t)row * K_ + c] = (h16)(val * r);
}

// ---------------- fused q/k/v projection GEMM (A reused 3x) ----------------
// q scaled by 1/sqrt(K)=1/16 at store; v stored transposed as v_t[b][n][t]
__global__ void k_qkv(const h16* __restrict__ gn,
                      const h16* __restrict__ qWh, const h16* __restrict__ kWh,
                      const h16* __restrict__ vWh,
                      h16* __restrict__ qh, h16* __restrict__ kh, h16* __restrict__ vt) {
  int wid  = (blockIdx.x * blockDim.x + threadIdx.x) >> 5;   // 4096 waves
  int lane = threadIdx.x & 31;
  int col = lane & 15, hs = lane >> 4;
  int rt = wid >> 4, nt = wid & 15;
  int row0 = rt << 4, n0 = nt << 4;
  const h16* abase = gn + (size_t)row0 * K_;
  v8f aq = {}, ak = {}, av = {};
#pragma unroll
  for (int c0 = 0; c0 < K_; c0 += 32) {
    v16h a = loadA(abase, K_, col, hs, c0);
    size_t boff = (size_t)(n0 + col) * K_ + c0 + hs * 16;
    aq = WMMA(a, loadB(qWh + boff), aq);
    ak = WMMA(a, loadB(kWh + boff), ak);
    av = WMMA(a, loadB(vWh + boff), av);
  }
  int b = row0 >> 11, tloc = row0 & (T_ - 1);
#pragma unroll
  for (int r = 0; r < 8; r++) {
    int me = r + 8 * hs;
    size_t idx = (size_t)(row0 + me) * K_ + n0 + col;
    qh[idx] = (h16)(aq[r] * 0.0625f);
    kh[idx] = (h16)(ak[r]);
    vt[(size_t)(b * K_ + n0 + col) * T_ + (tloc + me)] = (h16)(av[r]);
  }
}

// ---------------- streaming decayed attention ----------------
// Block = 4 waves sharing (b, t-tile of 16), each wave owns an n-tile of 64.
// k chunk (32 s-rows x 256) is identical for all 4 waves -> staged once per
// block into LDS by the Tensor Data Mover, double-buffered so the DMA of
// chunk i+1 overlaps the WMMA work of chunk i.
__global__ void k_attn(const h16* __restrict__ qh, const h16* __restrict__ kh,
                       const h16* __restrict__ vt, const float* __restrict__ dlogit,
                       h16* __restrict__ rh) {
  __shared__ __align__(128) h16 ldsk[2 * 32 * K_];   // 2 x 16KB k-chunk buffers
  __shared__ __align__(16)  h16 lds_s[4 * 16 * 40];  // per-wave score staging
  int wib  = threadIdx.x >> 5;
  int wid  = (blockIdx.x * blockDim.x + threadIdx.x) >> 5;   // 1024 waves
  int lane = threadIdx.x & 31;
  int col = lane & 15, hs = lane >> 4;
  int b  = wid >> 9;
  int tt = (wid >> 2) & 127;    // uniform across the block's 4 waves
  int nq = wid & 3;
  int t0 = tt << 4, n0 = nq << 6;

  const h16* qbase = qh + (size_t)(b * T_ + t0) * K_;
  const h16* kb = kh + (size_t)b * T_ * K_;
  const h16* vb = vt + (size_t)b * K_ * T_;
  v16h qa[8];
#pragma unroll
  for (int cc = 0; cc < 8; cc++) qa[cc] = loadA(qbase, K_, col, hs, cc * 32);

  float decay = 1.f / (1.f + __expf(-dlogit[0]));
  float ldec = __logf(decay);
  h16* ls = lds_s + wib * (16 * 40);

  int s_first = (t0 >> 5) << 5;   // earlier chunks have s <= t: weight == 0
  if (wib == 0) {
    tdm_load_k(kb + (size_t)s_first * K_, (unsigned)(uintptr_t)&ldsk[0]);
    __builtin_amdgcn_s_wait_tensorcnt(0);
  }
  __syncthreads();

  v8f acc0 = {}, acc1 = {}, acc2 = {}, acc3 = {};
  int ci = 0;
  for (int s0 = s_first; s0 < T_; s0 += 32, ci++) {
    const h16* lk = &ldsk[(ci & 1) * 32 * K_];
    if (wib == 0 && s0 + 32 < T_)     // kick DMA for next chunk into other buffer
      tdm_load_k(kb + (size_t)(s0 + 32) * K_,
                 (unsigned)(uintptr_t)&ldsk[((ci + 1) & 1) * 32 * K_]);
    __builtin_prefetch(vb + (size_t)(n0 + col) * T_ + s0 + 32, 0, 0);

    v8f sa = {}, sb = {};
#pragma unroll
    for (int cc = 0; cc < 8; cc++) {
      int c0 = cc * 32;
      sa = WMMA(qa[cc], loadB(lk + (size_t)(col)      * K_ + c0 + hs * 16), sa);
      sb = WMMA(qa[cc], loadB(lk + (size_t)(col + 16) * K_ + c0 + hs * 16), sb);
    }
    // decay mask + f32->f16, stage C-layout -> A-layout via per-wave LDS region
#pragma unroll
    for (int r = 0; r < 8; r++) {
      int me = r + 8 * hs;
      int t = t0 + me;
      int s_a = s0 + col;
      int s_b = s_a + 16;
      float wa = (s_a > t) ? __expf(ldec * (float)(s_a - t - 1)) : 0.f;
      float wb = (s_b > t) ? __expf(ldec * (float)(s_b - t - 1)) : 0.f;
      ls[me * 40 + col]      = (h16)(sa[r] * wa);
      ls[me * 40 + 16 + col] = (h16)(sb[r] * wb);
    }
    // same-wave LDS RAW: DS pipe is in-order per wave; compiler inserts waits
    v8h lo = *(const v8h*)(ls + col * 40 + hs * 16);
    v8h hi = *(const v8h*)(ls + col * 40 + 8 + hs * 16);
    v16h sA = __builtin_shufflevector(lo, hi, 0,1,2,3,4,5,6,7,8,9,10,11,12,13,14,15);
    acc0 = WMMA(sA, loadB(vb + (size_t)(n0 +  0 + col) * T_ + s0 + hs * 16), acc0);
    acc1 = WMMA(sA, loadB(vb + (size_t)(n0 + 16 + col) * T_ + s0 + hs * 16), acc1);
    acc2 = WMMA(sA, loadB(vb + (size_t)(n0 + 32 + col) * T_ + s0 + hs * 16), acc2);
    acc3 = WMMA(sA, loadB(vb + (size_t)(n0 + 48 + col) * T_ + s0 + hs * 16), acc3);

    if (wib == 0 && s0 + 32 < T_) __builtin_amdgcn_s_wait_tensorcnt(0);
    __syncthreads();   // next chunk resident; everyone done reading lk
  }
#pragma unroll
  for (int r = 0; r < 8; r++) {
    int me = r + 8 * hs;
    size_t base = (size_t)(b * T_ + t0 + me) * K_ + n0 + col;
    rh[base +  0] = (h16)acc0[r];
    rh[base + 16] = (h16)acc1[r];
    rh[base + 32] = (h16)acc2[r];
    rh[base + 48] = (h16)acc3[r];
  }
}

// ---------------- o-projection + residual ----------------
__global__ void k_oproj(const h16* __restrict__ rh, const h16* __restrict__ oWh,
                        const float* __restrict__ g, const float* __restrict__ mscale,
                        const float* __restrict__ oscale, float* __restrict__ g2) {
  int wid  = (blockIdx.x * blockDim.x + threadIdx.x) >> 5;   // 4096 waves
  int lane = threadIdx.x & 31;
  int col = lane & 15, hs = lane >> 4;
  int rt = wid >> 4, nt = wid & 15;
  int row0 = rt << 4, n0 = nt << 4;
  const h16* abase = rh + (size_t)row0 * K_;
  v8f acc = {};
#pragma unroll
  for (int c0 = 0; c0 < K_; c0 += 32) {
    v16h a = loadA(abase, K_, col, hs, c0);
    acc = WMMA(a, loadB(oWh + (size_t)(n0 + col) * K_ + c0 + hs * 16), acc);
  }
  float coef = mscale[0] * oscale[0];
#pragma unroll
  for (int r = 0; r < 8; r++) {
    int me = r + 8 * hs;
    size_t idx = (size_t)(row0 + me) * K_ + n0 + col;
    g2[idx] = g[idx] + coef * acc[r];
  }
}

// ---------------- down-projection + exact GELU ----------------
__global__ void k_down(const h16* __restrict__ gn2, const h16* __restrict__ dWh,
                       const float* __restrict__ bias, h16* __restrict__ hh) {
  int wid  = (blockIdx.x * blockDim.x + threadIdx.x) >> 5;   // 8192 waves
  int lane = threadIdx.x & 31;
  int col = lane & 15, hs = lane >> 4;
  int rt = wid >> 5, it = wid & 31;
  int row0 = rt << 4, i0 = it << 4;
  const h16* abase = gn2 + (size_t)row0 * K_;
  v8f acc = {};
#pragma unroll
  for (int c0 = 0; c0 < K_; c0 += 32) {
    v16h a = loadA(abase, K_, col, hs, c0);
    acc = WMMA(a, loadB(dWh + (size_t)(i0 + col) * K_ + c0 + hs * 16), acc);
  }
  float bn = bias[i0 + col];
#pragma unroll
  for (int r = 0; r < 8; r++) {
    int me = r + 8 * hs;
    float v = acc[r] + bn;
    float gl = 0.5f * v * (1.f + erff(v * 0.70710678118654752f));
    hh[(size_t)(row0 + me) * INNER_ + i0 + col] = (h16)gl;
  }
}

// ---------------- up-projection + scatter to write_indices ----------------
__global__ void k_up(const h16* __restrict__ hh, const h16* __restrict__ uWh,
                     const float* __restrict__ wscale, const int* __restrict__ widx,
                     float* __restrict__ out) {
  int wid  = (blockIdx.x * blockDim.x + threadIdx.x) >> 5;   // 4096 waves
  int lane = threadIdx.x & 31;
  int col = lane & 15, hs = lane >> 4;
  int rt = wid >> 4, nt = wid & 15;
  int row0 = rt << 4, n0 = nt << 4;
  const h16* abase = hh + (size_t)row0 * INNER_;
  v8f acc = {};
#pragma unroll
  for (int c0 = 0; c0 < INNER_; c0 += 32) {
    v16h a = loadA(abase, INNER_, col, hs, c0);
    acc = WMMA(a, loadB(uWh + (size_t)(n0 + col) * INNER_ + c0 + hs * 16), acc);
  }
  float sc = wscale[0] * 0.0625f;       // write_scale / sqrt(K)
  int vcol = widx[n0 + col];
#pragma unroll
  for (int r = 0; r < 8; r++) {
    int me = r + 8 * hs;
    out[(size_t)(row0 + me) * V_ + vcol] = acc[r] * sc;
  }
}

extern "C" void kernel_launch(void* const* d_in, const int* in_sizes, int n_in,
                              void* d_out, int out_size, void* d_ws, size_t ws_size,
                              hipStream_t stream) {
  (void)in_sizes; (void)n_in; (void)out_size; (void)ws_size;
  const float* x      = (const float*)d_in[0];
  const float* qW     = (const float*)d_in[1];
  const float* kW     = (const float*)d_in[2];
  const float* vW     = (const float*)d_in[3];
  const float* oW     = (const float*)d_in[4];
  const float* dlog   = (const float*)d_in[5];
  const float* oscale = (const float*)d_in[6];
  const float* downW  = (const float*)d_in[7];
  const float* upW    = (const float*)d_in[8];
  const float* bias   = (const float*)d_in[9];
  const float* mscale = (const float*)d_in[10];
  const float* wscale = (const float*)d_in[11];
  const int*   ridx   = (const int*)d_in[12];
  const int*   widx   = (const int*)d_in[13];

  char* p0 = (char*)d_ws;
  size_t off = 0;
  auto carve = [&](size_t bytes) -> void* {
    void* r = (void*)(p0 + off);
    off += (bytes + 255) & ~(size_t)255;
    return r;
  };
  h16* qWh  = (h16*)carve((size_t)K_ * K_ * 2);
  h16* kWh  = (h16*)carve((size_t)K_ * K_ * 2);
  h16* vWh  = (h16*)carve((size_t)K_ * K_ * 2);
  h16* oWh  = (h16*)carve((size_t)K_ * K_ * 2);
  h16* dWh  = (h16*)carve((size_t)INNER_ * K_ * 2);
  h16* uWh  = (h16*)carve((size_t)K_ * INNER_ * 2);
  float* g   = (float*)carve((size_t)BT_ * K_ * 4);
  h16*   gnh = (h16*)carve((size_t)BT_ * K_ * 2);
  h16*   qh  = (h16*)carve((size_t)BT_ * K_ * 2);
  h16*   kh  = (h16*)carve((size_t)BT_ * K_ * 2);
  h16*   vt  = (h16*)carve((size_t)BT_ * K_ * 2);
  h16*   rh  = (h16*)carve((size_t)BT_ * K_ * 2);
  float* g2  = (float*)carve((size_t)BT_ * K_ * 4);
  h16*   gn2 = (h16*)carve((size_t)BT_ * K_ * 2);
  h16*   hh  = (h16*)carve((size_t)BT_ * INNER_ * 2);

  // zero-fill output: the bandwidth-dominant step (~524 MB), non-temporal
  k_zero<<<4096, 256, 0, stream>>>((f32x4*)d_out, (size_t)BT_ * V_ / 4);

  // weight conversions (no transpose needed: W row-major == B-matrix n-major)
  k_f2h<<<64, 256, 0, stream>>>(qW, qWh, K_ * K_);
  k_f2h<<<64, 256, 0, stream>>>(kW, kWh, K_ * K_);
  k_f2h<<<64, 256, 0, stream>>>(vW, vWh, K_ * K_);
  k_f2h<<<64, 256, 0, stream>>>(oW, oWh, K_ * K_);
  k_f2h<<<128, 256, 0, stream>>>(downW, dWh, INNER_ * K_);
  k_f2h<<<128, 256, 0, stream>>>(upW, uWh, K_ * INNER_);

  k_gather_rms<<<BT_, 256, 0, stream>>>(x, ridx, g, gnh);
  k_qkv<<<1024, 128, 0, stream>>>(gnh, qWh, kWh, vWh, qh, kh, vt);
  k_attn<<<256, 128, 0, stream>>>(qh, kh, vt, dlog, rh);
  k_oproj<<<1024, 128, 0, stream>>>(rh, oWh, g, mscale, oscale, g2);
  k_rms<<<BT_, 256, 0, stream>>>(g2, gn2);
  k_down<<<2048, 128, 0, stream>>>(gn2, dWh, bias, hh);
  k_up<<<1024, 128, 0, stream>>>(hh, uWh, wscale, widx, (float*)d_out);
}
